// AttentionLayer_23622320128601
// MI455X (gfx1250) — matrix operand order repaired and checked by reading, not compiled, not run on hardware
//
#include <hip/hip_runtime.h>
#include <hip/hip_bf16.h>
#include <math.h>

typedef __bf16 bf16_t;
typedef __attribute__((ext_vector_type(16))) __bf16 v16bf;
typedef __attribute__((ext_vector_type(8)))  float  v8f;

#define LDSS 40   // LDS row stride in bf16 elements (32 data + 8 pad, keeps 16B alignment)

union FragU { uint4 u[2]; v16bf v; };

// EPI: 0 = h-projection (bias + tgt, *sqrt(.5), bf16 out)
//      1 = scores (mask -> -inf, f32 out)
//      2 = attn*V  (* sqrt(S), bf16 out)
//      3 = final   (bias + residual, *sqrt(.5), f32 out)
template<bool A_IS_BF16, bool B_TRANS, int EPI>
__global__ __launch_bounds__(256)
void gemm_wmma(const void* __restrict__ Abase, const float* __restrict__ Bbase,
               long long sA, long long sB,
               int M, int N, int K, int ldA, int ldB,
               const float* __restrict__ bias, const float* __restrict__ addf,
               const unsigned char* __restrict__ maskp, long long sMask,
               float* __restrict__ Df, long long sDf,
               bf16_t* __restrict__ Dbf, long long sDbf)
{
    // double-buffered tiles: A[128][32], B[128][32] (B stored n-major, k-contiguous)
    __shared__ alignas(16) bf16_t Abuf[2][128 * LDSS];
    __shared__ alignas(16) bf16_t Bbuf[2][128 * LDSS];

    const int bz    = blockIdx.z;
    const int tid   = threadIdx.x;
    const int lane  = tid & 31;
    const int wave  = tid >> 5;
    const int mBase = blockIdx.y * 128;
    const int nBase = blockIdx.x * 128;

    const float*  Af = (const float*)Abase  + (size_t)bz * sA;
    const bf16_t* Ab = (const bf16_t*)Abase + (size_t)bz * sA;
    const float*  Bg = Bbase + (size_t)bz * sB;

    // loader indices: A / non-transposed B -> 2 threads per row, 16 elems each
    const int arow = tid >> 1;
    const int akh  = (tid & 1) * 16;
    // transposed-B loader: 8 threads per k-row, 16 n each
    const int bk = tid >> 3;
    const int bn = (tid & 7) * 16;

    // wave tile: 2 waves in M (64 each), 4 waves in N (32 each)
    const int wm = (wave >> 2) * 64;
    const int wn = (wave & 3) * 32;
    const int lr = lane & 15;
    const int lh = (lane >> 4) & 1;

    // ---------- staging helpers ----------
    // bf16 A tile: CDNA5 async copy global->LDS (32B per thread), ASYNCcnt-tracked
    auto asyncA = [&](int k0, int buf) {
        unsigned lds = (unsigned)(uintptr_t)&Abuf[buf][arow * LDSS + akh];
        unsigned long long g =
            (unsigned long long)(uintptr_t)(Ab + (size_t)(mBase + arow) * ldA + k0 + akh);
        asm volatile("global_load_async_to_lds_b128 %0, %1, off\n\t"
                     "global_load_async_to_lds_b128 %0, %1, off offset:16"
                     :: "v"(lds), "v"(g) : "memory");
    };
    // f32 tile: load 16 floats into registers (issued early, consumed after WMMA)
    auto loadA_f32 = [&](int k0, float* r) {
        const float* src = Af + (size_t)(mBase + arow) * ldA + k0 + akh;
#pragma unroll
        for (int i = 0; i < 16; ++i) r[i] = src[i];
    };
    auto storeA_f32 = [&](int buf, const float* r) {
        alignas(16) bf16_t tmp[16];
#pragma unroll
        for (int i = 0; i < 16; ++i) tmp[i] = (bf16_t)r[i];
        ((uint4*)&Abuf[buf][arow * LDSS + akh])[0] = ((uint4*)tmp)[0];
        ((uint4*)&Abuf[buf][arow * LDSS + akh])[1] = ((uint4*)tmp)[1];
    };
    auto loadB = [&](int k0, float* r) {
        const float* src = B_TRANS ? (Bg + (size_t)(k0 + bk) * ldB + nBase + bn)
                                   : (Bg + (size_t)(nBase + arow) * ldB + k0 + akh);
#pragma unroll
        for (int i = 0; i < 16; ++i) r[i] = src[i];
    };
    auto storeB = [&](int buf, const float* r) {
        if (!B_TRANS) {
            alignas(16) bf16_t tmp[16];
#pragma unroll
            for (int i = 0; i < 16; ++i) tmp[i] = (bf16_t)r[i];
            ((uint4*)&Bbuf[buf][arow * LDSS + akh])[0] = ((uint4*)tmp)[0];
            ((uint4*)&Bbuf[buf][arow * LDSS + akh])[1] = ((uint4*)tmp)[1];
        } else {
#pragma unroll
            for (int i = 0; i < 16; ++i)
                Bbuf[buf][(bn + i) * LDSS + bk] = (bf16_t)r[i];
        }
    };

    v8f acc[4][2] = {};
    const int nk = K >> 5;

    // ---------- prologue: stage tile 0 into buffer 0 ----------
    {
        float r[16];
        if (A_IS_BF16) asyncA(0, 0);
        else { loadA_f32(0, r); storeA_f32(0, r); }
        float rb[16];
        loadB(0, rb); storeB(0, rb);
        if (A_IS_BF16) asm volatile("s_wait_asynccnt 0" ::: "memory");
    }
    __syncthreads();

    // ---------- main pipelined loop ----------
    for (int kt = 0; kt < nk; ++kt) {
        const int cur = kt & 1, nxt = cur ^ 1;
        const bool hasNext = (kt + 1) < nk;
        float ra[16], rb[16];

        // issue next-tile loads before the matrix ops (latency overlap)
        if (hasNext) {
            if (A_IS_BF16) asyncA((kt + 1) << 5, nxt);
            else           loadA_f32((kt + 1) << 5, ra);
            loadB((kt + 1) << 5, rb);
        }

        // ---- fragments + WMMA from current buffer ----
        FragU a[4], b[2];
#pragma unroll
        for (int i = 0; i < 4; ++i) {
            int ai = (wm + i * 16 + lr) * LDSS + lh * 8;
            a[i].u[0] = *(const uint4*)&Abuf[cur][ai];
            a[i].u[1] = *(const uint4*)&Abuf[cur][ai + 16];
        }
#pragma unroll
        for (int j = 0; j < 2; ++j) {
            int bi = (wn + j * 16 + lr) * LDSS + lh * 8;
            b[j].u[0] = *(const uint4*)&Bbuf[cur][bi];
            b[j].u[1] = *(const uint4*)&Bbuf[cur][bi + 16];
        }
#pragma unroll
        for (int i = 0; i < 4; ++i)
#pragma unroll
            for (int j = 0; j < 2; ++j)
                acc[i][j] = __builtin_amdgcn_wmma_f32_16x16x32_bf16(
                    false, a[i].v, false, b[j].v, (short)0, acc[i][j], false, false);

        // ---- commit next tile into the other buffer (no barrier needed: disjoint) ----
        if (hasNext) {
            if (!A_IS_BF16) storeA_f32(nxt, ra);
            storeB(nxt, rb);
            if (A_IS_BF16) asm volatile("s_wait_asynccnt 0" ::: "memory");
        }
        __syncthreads();
    }

    // ---------- epilogue ----------
    float*  DfB  = Df  + (size_t)bz * sDf;
    bf16_t* DbfB = Dbf + (size_t)bz * sDbf;
    const unsigned char* mrow = maskp ? (maskp + (size_t)bz * sMask) : nullptr;
    const float RS2 = 0.70710678118654752f;   // sqrt(0.5)
    const float SQS = 45.254833995939045f;    // sqrt(2048)

#pragma unroll
    for (int i = 0; i < 4; ++i) {
#pragma unroll
        for (int j = 0; j < 2; ++j) {
#pragma unroll
            for (int r = 0; r < 8; ++r) {
                int row = mBase + wm + i * 16 + lh * 8 + r;
                int col = nBase + wn + j * 16 + lr;
                float v = acc[i][j][r];
                if (EPI == 0) {
                    v = (v + bias[col] + addf[(size_t)row * N + col]) * RS2;
                    DbfB[(size_t)row * N + col] = (bf16_t)v;
                } else if (EPI == 1) {
                    if (mrow[col]) v = -__builtin_inff();
                    DfB[(size_t)row * N + col] = v;
                } else if (EPI == 2) {
                    DbfB[(size_t)row * N + col] = (bf16_t)(v * SQS);
                } else {
                    DfB[(size_t)row * N + col] =
                        (v + bias[col] + addf[(size_t)row * N + col]) * RS2;
                }
            }
        }
    }
}

// row-wise softmax over S=2048, in-place f32 + bf16 copy for next GEMM
__global__ __launch_bounds__(256)
void softmax_kernel(float* __restrict__ scores, bf16_t* __restrict__ attn_bf, int S)
{
    const int row = blockIdx.x;
    const int tid = threadIdx.x;
    float*  rp = scores  + (size_t)row * S;
    bf16_t* bp = attn_bf + (size_t)row * S;

    float v[8];
    float m = -__builtin_inff();
#pragma unroll
    for (int j = 0; j < 8; ++j) { v[j] = rp[tid + j * 256]; m = fmaxf(m, v[j]); }

    __shared__ float red[256];
    red[tid] = m; __syncthreads();
    for (int s = 128; s > 0; s >>= 1) {
        if (tid < s) red[tid] = fmaxf(red[tid], red[tid + s]);
        __syncthreads();
    }
    m = red[0]; __syncthreads();

    float sum = 0.f;
#pragma unroll
    for (int j = 0; j < 8; ++j) { v[j] = __expf(v[j] - m); sum += v[j]; }
    red[tid] = sum; __syncthreads();
    for (int s = 128; s > 0; s >>= 1) {
        if (tid < s) red[tid] += red[tid + s];
        __syncthreads();
    }
    sum = red[0];

    float inv = 1.0f / sum;
#pragma unroll
    for (int j = 0; j < 8; ++j) {
        float o = v[j] * inv;
        rp[tid + j * 256] = o;
        bp[tid + j * 256] = (bf16_t)o;
    }
}

extern "C" void kernel_launch(void* const* d_in, const int* in_sizes, int n_in,
                              void* d_out, int out_size, void* d_ws, size_t ws_size,
                              hipStream_t stream)
{
    constexpr int NB = 8, T = 2048, S = 2048, C = 1024, E = 1024;
    constexpr int M1 = NB * T;

    const float* x     = (const float*)d_in[0];
    const float* tgt   = (const float*)d_in[1];
    const float* enc_a = (const float*)d_in[2];
    const float* enc_b = (const float*)d_in[3];
    const unsigned char* mask = (const unsigned char*)d_in[4];
    const float* w_in  = (const float*)d_in[5];
    const float* b_in  = (const float*)d_in[6];
    const float* w_out = (const float*)d_in[7];
    const float* b_out = (const float*)d_in[8];

    float* out_f  = (float*)d_out;                       // [B,T,C]
    float* attn_f = out_f + (size_t)NB * T * C;          // [B,T,S]

    bf16_t* h_bf    = (bf16_t*)d_ws;                     // [B,T,E]
    bf16_t* attn_bf = h_bf    + (size_t)NB * T * E;      // [B,T,S]
    bf16_t* mid_bf  = attn_bf + (size_t)NB * T * S;      // [B,T,E]

    // 1) h = (x @ w_in^T + b_in + tgt) * sqrt(.5)   [M1 x E x C]
    gemm_wmma<false, false, 0><<<dim3(E / 128, M1 / 128, 1), 256, 0, stream>>>(
        (const void*)x, w_in, 0, 0, M1, E, C, C, C,
        b_in, tgt, nullptr, 0, nullptr, 0, h_bf, 0);

    // 2) scores = h @ enc_a, mask applied   [T x S x E] per batch
    gemm_wmma<true, true, 1><<<dim3(S / 128, T / 128, NB), 256, 0, stream>>>(
        (const void*)h_bf, enc_a, (long long)T * E, (long long)E * S,
        T, S, E, E, S,
        nullptr, nullptr, mask, S, attn_f, (long long)T * S, nullptr, 0);

    // 3) softmax rows, in-place + bf16 copy
    softmax_kernel<<<NB * T, 256, 0, stream>>>(attn_f, attn_bf, S);

    // 4) mid = attn @ enc_b * sqrt(S)   [T x E x S] per batch
    gemm_wmma<true, true, 2><<<dim3(E / 128, T / 128, NB), 256, 0, stream>>>(
        (const void*)attn_bf, enc_b, (long long)T * S, (long long)S * E,
        T, E, S, S, E,
        nullptr, nullptr, nullptr, 0, nullptr, 0, mid_bf, (long long)T * E);

    // 5) out = (mid @ w_out^T + b_out + x) * sqrt(.5)   [M1 x C x E]
    gemm_wmma<true, false, 3><<<dim3(C / 128, M1 / 128, 1), 256, 0, stream>>>(
        (const void*)mid_bf, w_out, 0, 0, M1, C, E, E, E,
        b_out, x, nullptr, 0, out_f, 0, nullptr, 0);
}
